// GATMLPLayer_1486058684460
// MI455X (gfx1250) — compile-verified, matrix operand order checked by reading, not run on hardware
//
#include <hip/hip_runtime.h>
#include <hip/hip_bf16.h>

// ---------------------------------------------------------------------------
// GATv2Conv (H=5, C=64, mean-aggregated heads) + residual LN + MLP + LN
// Optimized for MI455X (gfx1250, wave32, WMMA 16x16x32 f16, 192MB L2).
//
// Strategy:
//  * All GEMMs (xl/xr = x@W, em = ea@We fused into edge kernel, MLP) run on
//    v_wmma_f32_16x16x32_f16 (f16 in, f32 accumulate). K=64 -> 2 WMMA steps.
//  * em ([850k,320] = 1.09GB) is NEVER materialized: each wave computes a
//    16-edge x 16-col tile in registers and fuses leaky_relu + att-dot.
//  * xl/xr (64MB each) both fit in the 192MB L2 -> edge gathers are L2-bound.
//  * WMMA(f16) is TRANS-tracked and co-executes with VALU, so the edge kernel
//    is VALU-limited: cross-lane reductions are hoisted per-head (4x fewer
//    shuffles) and per-(edge,head) LDS atomics replaced by unique-writer
//    stores.
//  * Scatter softmax: atomic-max (order-preserving uint map) + atomic-add.
// ---------------------------------------------------------------------------

#define N_NODES 50000
#define E_EDGES 800000
#define EN_TOT  850000   // E + N (self loops)
#define NG      53125    // EN_TOT / 16 edge groups
#define HC      320      // H * C
#define NHEAD   5

typedef __attribute__((ext_vector_type(16))) _Float16 v16h;
typedef __attribute__((ext_vector_type(8)))  _Float16 v8h;
typedef __attribute__((ext_vector_type(8)))  float    v8f;

__device__ __forceinline__ v8f wmma_f16(v16h a, v16h b, v8f c) {
    return __builtin_amdgcn_wmma_f32_16x16x32_f16(
        false, a, false, b, (short)0, c, false, false);
}

// Build an A-fragment (16x32 f16) per documented CDNA5 layout:
// lane<16: K = {0..7, 16..23}+kbase ; lane>=16: K = {8..15, 24..31}+kbase
// rowp points at this lane's row (M = lane&15).
__device__ __forceinline__ v16h load_a16(const _Float16* rowp, int kbase, int lane) {
    const int off = (lane & 16) ? 8 : 0;
    union { v16h v; v8h h[2]; } u;
    u.h[0] = *(const v8h*)(rowp + kbase + off);
    u.h[1] = *(const v8h*)(rowp + kbase + 16 + off);
    return u.v;
}

// Order-preserving float<->uint map for atomicMax on floats (any sign).
__device__ __forceinline__ unsigned f2ord(float f) {
    unsigned u = __float_as_uint(f);
    return (u >> 31) ? ~u : (u ^ 0x80000000u);
}
__device__ __forceinline__ float ord2f(unsigned u) {
    return (u >> 31) ? __uint_as_float(u ^ 0x80000000u) : __uint_as_float(~u);
}

// Sum across each 16-lane half of the wave (edges live per-half in C/D layout).
__device__ __forceinline__ float hreduce16(float x) {
    x += __shfl_xor(x, 1, 32);
    x += __shfl_xor(x, 2, 32);
    x += __shfl_xor(x, 4, 32);
    x += __shfl_xor(x, 8, 32);
    return x;
}

// ---------------------------------------------------------------------------
__global__ void k_init(float* cnt, float* sumattr, float* den, float* acc,
                       unsigned* mxu) {
    int i = blockIdx.x * blockDim.x + threadIdx.x;
    if (i < N_NODES * 64) { sumattr[i] = 0.f; acc[i] = 0.f; }
    if (i < N_NODES * NHEAD) { den[i] = 0.f; mxu[i] = 0x007FFFFFu; } // ord(-inf)
    if (i < N_NODES) cnt[i] = 0.f;
}

__global__ void k_count(const long long* __restrict__ ei,
                        const float* __restrict__ rel,
                        float* cnt, float* sumattr) {
    long long i = (long long)blockIdx.x * blockDim.x + threadIdx.x;
    if (i >= (long long)E_EDGES * 64) return;
    int e = (int)(i >> 6), c = (int)(i & 63);
    int d = (int)ei[E_EDGES + e];
    atomicAdd(&sumattr[(size_t)d * 64 + c], rel[i]);
    if (c == 0) atomicAdd(&cnt[d], 1.f);
}

__global__ void k_cast(const float* __restrict__ src, _Float16* __restrict__ dst,
                       long long total) {
    long long i = (long long)blockIdx.x * blockDim.x + threadIdx.x;
    if (i < total) dst[i] = (_Float16)src[i];
}

// self-loop edge_attr = mean of incoming edge_attr
__global__ void k_loopattr(const float* __restrict__ sumattr,
                           const float* __restrict__ cnt,
                           _Float16* __restrict__ ea_node) {
    int i = blockIdx.x * blockDim.x + threadIdx.x;
    if (i >= N_NODES * 64) return;
    int n = i >> 6;
    ea_node[i] = (_Float16)(sumattr[i] / fmaxf(cnt[n], 1.f));
}

// Pre-swizzle weight matrix W[K][ncols] (row major) into B-fragment layout
// (B^T in A layout): F[(tn*nkb+kb)*32 + lane][elem]
__global__ void k_wfrag(const float* __restrict__ W, _Float16* __restrict__ F,
                        int ncols, int nk) {
    int i = blockIdx.x * blockDim.x + threadIdx.x;
    int nkb = nk >> 5;
    int total = (ncols >> 4) * nkb * 32 * 16;
    if (i >= total) return;
    int elem = i & 15;
    int lane = (i >> 4) & 31;
    int j = i >> 9;                 // tn*nkb + kb
    int kb = j % nkb, tn = j / nkb;
    int n = tn * 16 + (lane & 15);
    int off = (lane & 16) ? 8 : 0;
    int k = kb * 32 + ((elem < 8) ? (off + elem) : (16 + off + (elem - 8)));
    F[i] = (_Float16)W[(size_t)k * ncols + n];
}

// xl = x@Wl + bl ; xr = x@Wr + br  -- one wave per 16x16 output tile.
__global__ void k_gemm_xlr(const _Float16* __restrict__ xh,
                           const _Float16* __restrict__ Fl,
                           const _Float16* __restrict__ Fr,
                           const float* __restrict__ bl,
                           const float* __restrict__ br,
                           float* __restrict__ xl, float* __restrict__ xr) {
    int w = threadIdx.x >> 5, lane = threadIdx.x & 31;
    int t = blockIdx.x * 8 + w;
    if (t >= 125000) return;              // 3125 Mtiles * 20 Ntiles * 2
    int which = t / 62500, r = t % 62500;
    int mt = r / 20, nt = r % 20;
    const _Float16* F = which ? Fr : Fl;
    const float* bias = which ? br : bl;
    float* out = which ? xr : xl;

    const _Float16* rowp = xh + (size_t)(mt * 16 + (lane & 15)) * 64;
    v16h a0 = load_a16(rowp, 0, lane);
    v16h a1 = load_a16(rowp, 32, lane);
    v16h b0 = *(const v16h*)(F + ((size_t)(nt * 2 + 0) * 32 + lane) * 16);
    v16h b1 = *(const v16h*)(F + ((size_t)(nt * 2 + 1) * 32 + lane) * 16);
    v8f c = {};
    c = wmma_f16(a0, b0, c);
    c = wmma_f16(a1, b1, c);

    int col = nt * 16 + (lane & 15);
    float bv = bias[col];
    int rbase = mt * 16 + ((lane & 16) ? 8 : 0);
#pragma unroll
    for (int v = 0; v < 8; ++v)
        out[(size_t)(rbase + v) * HC + col] = c[v] + bv;
}

// Fused edge kernel: em tile (WMMA) + xl[src]+xr[dst] gather + leaky_relu +
// att dot; per-head lane-partial accumulation, ONE cross-lane reduce per
// (head, edge-slot); scatter atomic-max per (dst,head).
__global__ void k_logits(const _Float16* __restrict__ eah,
                         const _Float16* __restrict__ Fe,
                         const long long* __restrict__ ei,
                         const float* __restrict__ xl,
                         const float* __restrict__ xr,
                         const float* __restrict__ att,
                         float* __restrict__ logits, unsigned* mxu) {
    __shared__ int   s_src[8][16], s_dst[8][16];
    __shared__ float s_lg[8][16][NHEAD];
    int w = threadIdx.x >> 5, lane = threadIdx.x & 31;
    int g = blockIdx.x * 8 + w;
    bool active = g < NG;

    if (active && lane < 16) {
        int e = g * 16 + lane, s, d;
        if (e < E_EDGES) { s = (int)ei[e]; d = (int)ei[E_EDGES + e]; }
        else             { s = d = e - E_EDGES; }
        s_src[w][lane] = s; s_dst[w][lane] = d;
    }
    __syncthreads();

    if (active) {
        const _Float16* rowp = eah + (size_t)(g * 16 + (lane & 15)) * 64;
        v16h a0 = load_a16(rowp, 0, lane);
        v16h a1 = load_a16(rowp, 32, lane);
        int m0 = (lane & 16) ? 8 : 0;
        int sr[8], dr[8];
#pragma unroll
        for (int v = 0; v < 8; ++v) { sr[v] = s_src[w][v + m0]; dr[v] = s_dst[w][v + m0]; }

        for (int h = 0; h < NHEAD; ++h) {
            float accv[8];
#pragma unroll
            for (int v = 0; v < 8; ++v) accv[v] = 0.f;
            for (int q = 0; q < 4; ++q) {        // 4 col-tiles per head (C=64)
                int nt = h * 4 + q;
                v16h b0 = *(const v16h*)(Fe + ((size_t)(nt * 2 + 0) * 32 + lane) * 16);
                v16h b1 = *(const v16h*)(Fe + ((size_t)(nt * 2 + 1) * 32 + lane) * 16);
                v8f c = {};
                c = wmma_f16(a0, b0, c);
                c = wmma_f16(a1, b1, c);
                int col = nt * 16 + (lane & 15);
                float av = att[col];
#pragma unroll
                for (int v = 0; v < 8; ++v) {
                    float s = c[v] + xl[(size_t)sr[v] * HC + col]
                                   + xr[(size_t)dr[v] * HC + col];
                    float m = (s > 0.f) ? s : 0.2f * s;      // leaky_relu
                    accv[v] += m * av;
                }
            }
#pragma unroll
            for (int v = 0; v < 8; ++v) {
                float r = hreduce16(accv[v]);
                if ((lane & 15) == 0)            // unique writer per (edge,head)
                    s_lg[w][v + m0][h] = r;
            }
        }
    }
    __syncthreads();

    if (active) {
        for (int i = lane; i < 16 * NHEAD; i += 32) {
            int e = i / NHEAD, h = i - e * NHEAD;
            float lg = s_lg[w][e][h];
            int ge = g * 16 + e;
            logits[(size_t)ge * NHEAD + h] = lg;
            int d = s_dst[w][e];
            atomicMax(&mxu[(size_t)d * NHEAD + h], f2ord(lg));
        }
    }
}

__global__ void k_passB(const float* __restrict__ logits,
                        const long long* __restrict__ ei,
                        const unsigned* __restrict__ mxu,
                        float* __restrict__ z, float* den) {
    long long i = (long long)blockIdx.x * blockDim.x + threadIdx.x;
    if (i >= (long long)EN_TOT * NHEAD) return;
    int e = (int)(i / NHEAD), h = (int)(i - (long long)e * NHEAD);
    int d = (e < E_EDGES) ? (int)ei[E_EDGES + e] : (e - E_EDGES);
    float zz = __expf(logits[i] - ord2f(mxu[(size_t)d * NHEAD + h]));
    z[i] = zz;
    atomicAdd(&den[(size_t)d * NHEAD + h], zz);
}

// Weighted scatter: acc[dst][c] += (1/H) * sum_h alpha[e,h] * xl[src][h*64+c]
__global__ void k_passC(const float* __restrict__ z,
                        const float* __restrict__ den,
                        const long long* __restrict__ ei,
                        const float* __restrict__ xl, float* acc) {
    int w = threadIdx.x >> 5, lane = threadIdx.x & 31;
    int e = blockIdx.x * 8 + w;
    if (e >= EN_TOT) return;
    int s, d;
    if (e < E_EDGES) { s = (int)ei[e]; d = (int)ei[E_EDGES + e]; }
    else             { s = d = e - E_EDGES; }
    float al[NHEAD];
#pragma unroll
    for (int h = 0; h < NHEAD; ++h)
        al[h] = z[(size_t)e * NHEAD + h] / den[(size_t)d * NHEAD + h];
    const float* xls = xl + (size_t)s * HC;
#pragma unroll
    for (int t = 0; t < 2; ++t) {
        int c = lane + t * 32;
        float sum = 0.f;
#pragma unroll
        for (int h = 0; h < NHEAD; ++h) sum += al[h] * xls[h * 64 + c];
        atomicAdd(&acc[(size_t)d * 64 + c], 0.2f * sum);   // 1/H fold
    }
}

// LayerNorm over 64 features, one wave per node (2 cols per lane).
// out = LN(a [+ b] [+ cb]) * g + beta ; optional f16 copy for WMMA input.
__global__ void k_ln(const float* __restrict__ a, const float* __restrict__ b,
                     const float* __restrict__ cb, const float* __restrict__ g,
                     const float* __restrict__ beta, float* __restrict__ out,
                     _Float16* __restrict__ outh) {
    int w = threadIdx.x >> 5, lane = threadIdx.x & 31;
    int n = blockIdx.x * 8 + w;
    if (n >= N_NODES) return;
    size_t base = (size_t)n * 64;
    int c0 = lane, c1 = lane + 32;
    float v0 = a[base + c0], v1 = a[base + c1];
    if (b)  { v0 += b[base + c0]; v1 += b[base + c1]; }
    if (cb) { v0 += cb[c0]; v1 += cb[c1]; }
    float s = v0 + v1, q = v0 * v0 + v1 * v1;
#pragma unroll
    for (int o = 1; o < 32; o <<= 1) {
        s += __shfl_xor(s, o, 32);
        q += __shfl_xor(q, o, 32);
    }
    float m   = s * (1.f / 64.f);
    float var = q * (1.f / 64.f) - m * m;
    float r   = rsqrtf(var + 1e-5f);
    float o0 = (v0 - m) * r * g[c0] + beta[c0];
    float o1 = (v1 - m) * r * g[c1] + beta[c1];
    out[base + c0] = o0; out[base + c1] = o1;
    if (outh) { outh[base + c0] = (_Float16)o0; outh[base + c1] = (_Float16)o1; }
}

// Fused MLP: t = x2@fc_w + fc_b (16 WMMA); LDS relayout f32->f16;
// h = t@fc1_w + fc1_b (16 WMMA); y = x2 + h.
__global__ void k_mlp(const _Float16* __restrict__ x2h,
                      const _Float16* __restrict__ Ffc,
                      const _Float16* __restrict__ Ffc1,
                      const float* __restrict__ fcb,
                      const float* __restrict__ fc1b,
                      const float* __restrict__ x2, float* __restrict__ y) {
    __shared__ _Float16 s_t[8][16][128];
    int w = threadIdx.x >> 5, lane = threadIdx.x & 31;
    int mt = blockIdx.x * 8 + w;
    bool active = mt < 3125;

    if (active) {
        const _Float16* rowp = x2h + (size_t)(mt * 16 + (lane & 15)) * 64;
        v16h a0 = load_a16(rowp, 0, lane);
        v16h a1 = load_a16(rowp, 32, lane);
        int rb = (lane & 16) ? 8 : 0;
        for (int nt = 0; nt < 8; ++nt) {
            v16h b0 = *(const v16h*)(Ffc + ((size_t)(nt * 2 + 0) * 32 + lane) * 16);
            v16h b1 = *(const v16h*)(Ffc + ((size_t)(nt * 2 + 1) * 32 + lane) * 16);
            v8f c = {};
            c = wmma_f16(a0, b0, c);
            c = wmma_f16(a1, b1, c);
            int col = nt * 16 + (lane & 15);
            float bv = fcb[col];
#pragma unroll
            for (int v = 0; v < 8; ++v)
                s_t[w][rb + v][col] = (_Float16)(c[v] + bv);
        }
    }
    __syncthreads();

    if (active) {
        const _Float16* rowp = &s_t[w][lane & 15][0];
        v16h aa[4];
#pragma unroll
        for (int kb = 0; kb < 4; ++kb) aa[kb] = load_a16(rowp, kb * 32, lane);
        int rb = (lane & 16) ? 8 : 0;
        for (int nt = 0; nt < 4; ++nt) {
            v8f c = {};
#pragma unroll
            for (int kb = 0; kb < 4; ++kb) {
                v16h b = *(const v16h*)(Ffc1 + ((size_t)(nt * 4 + kb) * 32 + lane) * 16);
                c = wmma_f16(aa[kb], b, c);
            }
            int col = nt * 16 + (lane & 15);
            float bv = fc1b[col];
#pragma unroll
            for (int v = 0; v < 8; ++v) {
                size_t idx = (size_t)(mt * 16 + rb + v) * 64 + col;
                y[idx] = x2[idx] + c[v] + bv;       // residual x2 + h
            }
        }
    }
}

// ---------------------------------------------------------------------------
extern "C" void kernel_launch(void* const* d_in, const int* in_sizes, int n_in,
                              void* d_out, int out_size, void* d_ws, size_t ws_size,
                              hipStream_t stream) {
    (void)in_sizes; (void)n_in; (void)out_size; (void)ws_size;
    const float*     features = (const float*)d_in[0];
    const long long* ei       = (const long long*)d_in[1];   // [2,E] int64
    const float*     rel      = (const float*)d_in[2];
    const float *Wl = (const float*)d_in[3],  *bl = (const float*)d_in[4];
    const float *Wr = (const float*)d_in[5],  *br = (const float*)d_in[6];
    const float *We = (const float*)d_in[7];
    const float *att = (const float*)d_in[8], *cbias = (const float*)d_in[9];
    const float *ln_g = (const float*)d_in[10], *ln_b = (const float*)d_in[11];
    const float *ln1_g = (const float*)d_in[12], *ln1_b = (const float*)d_in[13];
    const float *fc_w = (const float*)d_in[14], *fc_b = (const float*)d_in[15];
    const float *fc1_w = (const float*)d_in[16], *fc1_b = (const float*)d_in[17];
    float* out = (float*)d_out;

    char* p = (char*)d_ws;
    auto alloc = [&](size_t bytes) {
        char* r = p; p += (bytes + 255) & ~(size_t)255; return r;
    };
    _Float16* xh      = (_Float16*)alloc((size_t)N_NODES * 64 * 2);
    _Float16* eah     = (_Float16*)alloc((size_t)EN_TOT * 64 * 2);
    float*    xl      = (float*)alloc((size_t)N_NODES * HC * 4);
    float*    xr      = (float*)alloc((size_t)N_NODES * HC * 4);
    float*    logits  = (float*)alloc((size_t)EN_TOT * NHEAD * 4);
    float*    zbuf    = (float*)alloc((size_t)EN_TOT * NHEAD * 4);
    unsigned* mxu     = (unsigned*)alloc((size_t)N_NODES * NHEAD * 4);
    float*    den     = (float*)alloc((size_t)N_NODES * NHEAD * 4);
    float*    cnt     = (float*)alloc((size_t)N_NODES * 4);
    float*    sumattr = (float*)alloc((size_t)N_NODES * 64 * 4);
    float*    acc     = (float*)alloc((size_t)N_NODES * 64 * 4);
    float*    x2      = (float*)alloc((size_t)N_NODES * 64 * 4);
    _Float16* x2h     = (_Float16*)alloc((size_t)N_NODES * 64 * 2);
    float*    ybuf    = (float*)alloc((size_t)N_NODES * 64 * 4);
    _Float16* Fl   = (_Float16*)alloc(20 * 2 * 32 * 16 * 2);
    _Float16* Fr   = (_Float16*)alloc(20 * 2 * 32 * 16 * 2);
    _Float16* Fe   = (_Float16*)alloc(20 * 2 * 32 * 16 * 2);
    _Float16* Ffc  = (_Float16*)alloc(8 * 2 * 32 * 16 * 2);
    _Float16* Ffc1 = (_Float16*)alloc(4 * 4 * 32 * 16 * 2);

    // 1) zero accumulators / init maxima
    k_init<<<(N_NODES * 64 + 255) / 256, 256, 0, stream>>>(cnt, sumattr, den, acc, mxu);
    // 2) weight fragment swizzles (tiny)
    k_wfrag<<<(20480 + 255) / 256, 256, 0, stream>>>(Wl, Fl, HC, 64);
    k_wfrag<<<(20480 + 255) / 256, 256, 0, stream>>>(Wr, Fr, HC, 64);
    k_wfrag<<<(20480 + 255) / 256, 256, 0, stream>>>(We, Fe, HC, 64);
    k_wfrag<<<(8192 + 255) / 256, 256, 0, stream>>>(fc_w, Ffc, 128, 64);
    k_wfrag<<<(8192 + 255) / 256, 256, 0, stream>>>(fc1_w, Ffc1, 64, 128);
    // 3) f16 casts
    k_cast<<<(int)(((long long)N_NODES * 64 + 255) / 256), 256, 0, stream>>>(
        features, xh, (long long)N_NODES * 64);
    k_cast<<<(int)(((long long)E_EDGES * 64 + 255) / 256), 256, 0, stream>>>(
        rel, eah, (long long)E_EDGES * 64);
    // 4) self-loop attr mean
    k_count<<<(int)(((long long)E_EDGES * 64 + 255) / 256), 256, 0, stream>>>(
        ei, rel, cnt, sumattr);
    k_loopattr<<<(N_NODES * 64 + 255) / 256, 256, 0, stream>>>(
        sumattr, cnt, eah + (size_t)E_EDGES * 64);
    // 5) xl / xr GEMMs (WMMA)
    k_gemm_xlr<<<15625, 256, 0, stream>>>(xh, Fl, Fr, bl, br, xl, xr);
    // 6) fused em + attention logits (WMMA) + scatter max
    k_logits<<<(NG + 7) / 8, 256, 0, stream>>>(eah, Fe, ei, xl, xr, att, logits, mxu);
    // 7) softmax numerator + denominators
    k_passB<<<(int)(((long long)EN_TOT * NHEAD + 255) / 256), 256, 0, stream>>>(
        logits, ei, mxu, zbuf, den);
    // 8) weighted scatter aggregation (heads folded)
    k_passC<<<(EN_TOT + 7) / 8, 256, 0, stream>>>(zbuf, den, ei, xl, acc);
    // 9) x2 = LN(features + conv)
    k_ln<<<(N_NODES + 7) / 8, 256, 0, stream>>>(features, acc, cbias, ln_g, ln_b, x2, x2h);
    // 10) y = x2 + MLP(x2)  (WMMA)
    k_mlp<<<(3125 + 7) / 8, 256, 0, stream>>>(x2h, Ffc, Ffc1, fc_b, fc1_b, x2, ybuf);
    // 11) out = LN(y)
    k_ln<<<(N_NODES + 7) / 8, 256, 0, stream>>>(ybuf, nullptr, nullptr, ln1_g, ln1_b,
                                                out, nullptr);
}
